// PAEDSR_8598524526883
// MI455X (gfx1250) — matrix operand order repaired and checked by reading, not compiled, not run on hardware
//
#include <hip/hip_runtime.h>
#include <math.h>

// ---------------- problem constants ----------------
#define Cc    256
#define CRr   32
#define Hh    40
#define Wwid  40
#define HW    1600          // H*W
#define LTOT  5280          // sum of hh*ww over 5 scales
#define K1    288           // CR*9
#define K2    2304          // C*9
#define SMS   10.0f         // softmax scale
#define EPSc  1e-4f

// scale segments: dims 40,36,32,28,24 ; offsets 0,1600,2896,3920,4704
__device__ __forceinline__ void seg_of(int l, int& off, int& hh) {
    if (l < 1600)       { off = 0;    hh = 40; }
    else if (l < 2896)  { off = 1600; hh = 36; }
    else if (l < 3920)  { off = 2896; hh = 32; }
    else if (l < 4704)  { off = 3920; hh = 28; }
    else                { off = 4704; hh = 24; }
}

// ---------------- WMMA types ----------------
typedef __attribute__((ext_vector_type(16))) _Float16 v16h;
typedef __attribute__((ext_vector_type(8)))  _Float16 v8h;
typedef __attribute__((ext_vector_type(8)))  float    v8f;

union H16 { v16h v; v8h h[2]; };

// ---------------- CDNA5 async-copy helpers ----------------
__device__ __forceinline__ void async_load_b64_to_lds(unsigned lds_off,
                                                      const _Float16* gaddr) {
    unsigned long long ga = (unsigned long long)(const void*)gaddr;
    asm volatile("global_load_async_to_lds_b64 %0, %1, off"
                 :: "v"(lds_off), "v"(ga) : "memory");
}

__device__ __forceinline__ void wait_async0() {
#if __has_builtin(__builtin_amdgcn_s_wait_asynccnt)
    __builtin_amdgcn_s_wait_asynccnt(0);
#else
    asm volatile("s_wait_asynccnt 0x0" ::: "memory");
#endif
}

// ---------------------------------------------------------------------------
// Block-tiled f16 GEMM, f32 accumulation, double-buffered async LDS staging:
//   C[M,N] = A[M,K] * B[N,K]^T      (A,B row-major, K contiguous, ld == K)
// Waves arranged WM x WN; each wave computes TWO 16x16 tiles (32 N-columns)
// sharing one A fragment. Macro tile = (WM*16) x (WN*32).
// K-panels of 32 staged with GLOBAL_LOAD_ASYNC_TO_LDS (ASYNCcnt) into the
// ping/pong buffer while WMMA consumes the other; one barrier per K-step.
// All control flow is wave-uniform (EXEC all-ones for WMMA). Grids exact:
// M % (WM*16) == 0, N % (WN*32) == 0, K % 32 == 0.
// ---------------------------------------------------------------------------
template<int WM, int WN>
__global__ __launch_bounds__(WM* WN * 32) void gemm_f16_wmma_lds(
    const _Float16* __restrict__ A, const _Float16* __restrict__ B,
    float* __restrict__ C, int M, int N, int K)
{
    constexpr int BM = WM * 16, BN = WN * 32, T = WM * WN * 32;
    constexpr int QA = BM * 8, QB = BN * 8;      // 8-byte qwords per K-panel
    constexpr int IA = QA / T, IB = QB / T;      // async ops per thread (exact)
    __shared__ __align__(16) _Float16 sA[2 * BM * 32];
    __shared__ __align__(16) _Float16 sB[2 * BN * 32];

    const int tid  = threadIdx.x;
    const int lane = tid & 31;
    const int wave = tid >> 5;
    const int wm = wave / WN, wn = wave % WN;

    const int nBlocksN = N / BN;
    const int m0 = (blockIdx.x / nBlocksN) * BM;
    const int n0 = (blockIdx.x % nBlocksN) * BN;

    const int h   = lane >> 4;              // half-wave select
    const int rA  = wm * 16 + (lane & 15);  // local A row of this lane's frag
    const int cB0 = wn * 32 + (lane & 15);  // first local B column

    const unsigned sA0 = (unsigned)(size_t)&sA[0];
    const unsigned sB0 = (unsigned)(size_t)&sB[0];

    const int nk = K >> 5;

    // ---- prologue: stage K-panel 0 into buffer 0 (uniform trip counts) ----
    #pragma unroll
    for (int i = 0; i < IA; ++i) {
        int q = tid + i * T;                 // qword -> row q/8, k (q%8)*4
        async_load_b64_to_lds(sA0 + (unsigned)q * 8u,
                              A + (size_t)(m0 + (q >> 3)) * K + ((q & 7) << 2));
    }
    #pragma unroll
    for (int i = 0; i < IB; ++i) {
        int q = tid + i * T;
        async_load_b64_to_lds(sB0 + (unsigned)q * 8u,
                              B + (size_t)(n0 + (q >> 3)) * K + ((q & 7) << 2));
    }

    v8f acc0 = {}, acc1 = {};
    for (int kk = 0; kk < nk; ++kk) {
        wait_async0();                       // own panel-kk async ops done
        __syncthreads();                     // everyone's panel-kk visible

        const int cur = kk & 1;
        if (kk + 1 < nk) {                   // stage next panel into other buf
            const int nxt = cur ^ 1;
            const int k0n = (kk + 1) << 5;
            #pragma unroll
            for (int i = 0; i < IA; ++i) {
                int q = tid + i * T;
                async_load_b64_to_lds(
                    sA0 + (unsigned)(nxt * BM * 64 + q * 8),
                    A + (size_t)(m0 + (q >> 3)) * K + k0n + ((q & 7) << 2));
            }
            #pragma unroll
            for (int i = 0; i < IB; ++i) {
                int q = tid + i * T;
                async_load_b64_to_lds(
                    sB0 + (unsigned)(nxt * BN * 64 + q * 8),
                    B + (size_t)(n0 + (q >> 3)) * K + k0n + ((q & 7) << 2));
            }
        }

        // ---- fragments from LDS (ds_load_b128) + 2 WMMAs sharing A ----
        const _Float16* pA = sA + cur * (BM * 32);
        const _Float16* pB = sB + cur * (BN * 32);
        H16 a, b0, b1;
        a.h[0]  = *(const v8h*)(pA + rA * 32 + 16 * h);        // K = 16h+e
        a.h[1]  = *(const v8h*)(pA + rA * 32 + 16 * h + 8);    // K = 16h+8+e
        b0.h[0] = *(const v8h*)(pB + cB0 * 32 + 16 * h);       // K = 16h+e
        b0.h[1] = *(const v8h*)(pB + cB0 * 32 + 16 * h + 8);
        b1.h[0] = *(const v8h*)(pB + (cB0 + 16) * 32 + 16 * h);
        b1.h[1] = *(const v8h*)(pB + (cB0 + 16) * 32 + 16 * h + 8);
        acc0 = __builtin_amdgcn_wmma_f32_16x16x32_f16(
            false, a.v, false, b0.v, (short)0, acc0, false, false);
        acc1 = __builtin_amdgcn_wmma_f32_16x16x32_f16(
            false, a.v, false, b1.v, (short)0, acc1, false, false);
    }

    // D layout: VGPR r -> row m0 + wm*16 + r + 8h; cols n0+cB0 and +16
    float* cp = C + (size_t)(m0 + wm * 16 + 8 * h) * N + n0 + cB0;
    #pragma unroll
    for (int r = 0; r < 8; ++r) {
        cp[(size_t)r * N]      = acc0[r];
        cp[(size_t)r * N + 16] = acc1[r];
    }
}

// ---------------------------------------------------------------------------
// conv1x1 + PReLU:  out[o,p] = prelu( sum_c W[o,c]*in[c,p] + b[o] )
// ---------------------------------------------------------------------------
__global__ __launch_bounds__(256) void conv1x1_prelu(
    const float* __restrict__ in, const float* __restrict__ Wt,
    const float* __restrict__ bias, const float* __restrict__ alpha,
    float* __restrict__ out, int O, int Cin, int P)
{
    int idx = blockIdx.x * blockDim.x + threadIdx.x;
    if (idx >= O * P) return;
    int o = idx / P, p = idx % P;
    float acc = bias[o];
    const float* wr = Wt + (size_t)o * Cin;
    for (int c = 0; c < Cin; ++c)
        acc = fmaf(wr[c], in[(size_t)c * P + p], acc);
    float a = alpha[0];
    out[(size_t)o * P + p] = (acc >= 0.f) ? acc : a * acc;
}

// ---------------------------------------------------------------------------
// scale-1.0 copy into the concatenated pyramid buffer  ref[C, LTOT]
// ---------------------------------------------------------------------------
__global__ __launch_bounds__(256) void copy_scale0(
    const float* __restrict__ xb, float* __restrict__ ref)
{
    int idx = blockIdx.x * blockDim.x + threadIdx.x;
    if (idx >= Cc * HW) return;
    int c = idx / HW, p = idx % HW;
    ref[(size_t)c * LTOT + p] = xb[idx];
}

// ---------------------------------------------------------------------------
// bicubic (Keys a=-0.5) downsample, boundary weights dropped + renormalized
// ---------------------------------------------------------------------------
__device__ __forceinline__ float keysw(float t) {
    t = fabsf(t);
    float t2 = t * t, t3 = t2 * t;
    if (t <= 1.f) return 1.5f * t3 - 2.5f * t2 + 1.f;
    if (t <  2.f) return -0.5f * t3 + 2.5f * t2 - 4.f * t + 2.f;
    return 0.f;
}

__global__ __launch_bounds__(256) void resize_cubic(
    const float* __restrict__ xb, float* __restrict__ ref, int hh, int ww, int off)
{
    int idx = blockIdx.x * blockDim.x + threadIdx.x;
    if (idx >= Cc * hh * ww) return;
    int c  = idx / (hh * ww);
    int r  = idx % (hh * ww);
    int oy = r / ww, ox = r % ww;
    float ry = (float)Hh   / (float)hh;
    float rx = (float)Wwid / (float)ww;
    float fy = (oy + 0.5f) * ry - 0.5f;
    float fx = (ox + 0.5f) * rx - 0.5f;
    int iy = (int)floorf(fy), ix = (int)floorf(fx);
    float wy[4], wx[4], sy = 0.f, sx = 0.f;
    #pragma unroll
    for (int m = 0; m < 4; ++m) {
        int yy = iy - 1 + m;
        float w = (yy >= 0 && yy < Hh) ? keysw(fy - (float)yy) : 0.f;
        wy[m] = w; sy += w;
    }
    #pragma unroll
    for (int n = 0; n < 4; ++n) {
        int xx = ix - 1 + n;
        float w = (xx >= 0 && xx < Wwid) ? keysw(fx - (float)xx) : 0.f;
        wx[n] = w; sx += w;
    }
    float acc = 0.f;
    #pragma unroll
    for (int m = 0; m < 4; ++m) {
        if (wy[m] == 0.f) continue;
        int yy = iy - 1 + m;
        float rowacc = 0.f;
        #pragma unroll
        for (int n = 0; n < 4; ++n) {
            if (wx[n] == 0.f) continue;
            int xx = ix - 1 + n;
            rowacc = fmaf(wx[n], xb[(size_t)c * HW + yy * Wwid + xx], rowacc);
        }
        acc = fmaf(wy[m], rowacc, acc);
    }
    ref[(size_t)c * LTOT + off + oy * ww + ox] = acc / (sy * sx);
}

// ---------------------------------------------------------------------------
// Wn[l, cr*9+d] = patch(refm) / max(||patch||, eps)   (f16, row-major, ld=288)
// ---------------------------------------------------------------------------
__global__ __launch_bounds__(256) void build_Wn(
    const float* __restrict__ refm, _Float16* __restrict__ Wn)
{
    int l = blockIdx.x * blockDim.x + threadIdx.x;
    if (l >= LTOT) return;
    int off, hh; seg_of(l, off, hh);
    int q = l - off, y = q / hh, x = q % hh;
    float ss = 0.f;
    for (int cr = 0; cr < CRr; ++cr)
        for (int d = 0; d < 9; ++d) {
            int py = y + d / 3 - 1, px = x + d % 3 - 1;
            float v = (py >= 0 && py < hh && px >= 0 && px < hh)
                          ? refm[(size_t)cr * LTOT + off + py * hh + px] : 0.f;
            ss = fmaf(v, v, ss);
        }
    float inv = 1.f / fmaxf(sqrtf(ss), EPSc);
    for (int cr = 0; cr < CRr; ++cr)
        for (int d = 0; d < 9; ++d) {
            int py = y + d / 3 - 1, px = x + d % 3 - 1;
            float v = (py >= 0 && py < hh && px >= 0 && px < hh)
                          ? refm[(size_t)cr * LTOT + off + py * hh + px] : 0.f;
            Wn[(size_t)l * K1 + cr * 9 + d] = (_Float16)(v * inv);
        }
}

// ---------------------------------------------------------------------------
// xp[p, cr*9+d] from match_base (f16, ld=288)
// ---------------------------------------------------------------------------
__global__ __launch_bounds__(256) void build_xp(
    const float* __restrict__ mb, _Float16* __restrict__ xp)
{
    int p = blockIdx.x * blockDim.x + threadIdx.x;
    if (p >= HW) return;
    int y = p / Wwid, x = p % Wwid;
    for (int cr = 0; cr < CRr; ++cr)
        for (int d = 0; d < 9; ++d) {
            int py = y + d / 3 - 1, px = x + d % 3 - 1;
            float v = (py >= 0 && py < Hh && px >= 0 && px < Wwid)
                          ? mb[(size_t)cr * HW + py * Wwid + px] : 0.f;
            xp[(size_t)p * K1 + cr * 9 + d] = (_Float16)v;
        }
}

// ---------------------------------------------------------------------------
// RallT[k2, l] = patch(base) transposed  (f16, ld=LTOT) : B-operand of GEMM2
// ---------------------------------------------------------------------------
__global__ __launch_bounds__(256) void build_RallT(
    const float* __restrict__ base, _Float16* __restrict__ RT)
{
    int idx = blockIdx.x * blockDim.x + threadIdx.x;
    if (idx >= K2 * LTOT) return;
    int k2 = idx / LTOT, l = idx % LTOT;
    int c = k2 / 9, d = k2 % 9;
    int off, hh; seg_of(l, off, hh);
    int q = l - off, y = q / hh, x = q % hh;
    int py = y + d / 3 - 1, px = x + d % 3 - 1;
    float v = (py >= 0 && py < hh && px >= 0 && px < hh)
                  ? base[(size_t)c * LTOT + off + py * hh + px] : 0.f;
    RT[(size_t)k2 * LTOT + l] = (_Float16)v;
}

// ---------------------------------------------------------------------------
// softmax over L (per query column p), writes transposed att as f16 [HW, LTOT]
// ---------------------------------------------------------------------------
__global__ __launch_bounds__(256) void softmax_col(
    const float* __restrict__ S, _Float16* __restrict__ attT)
{
    int p = blockIdx.x;
    __shared__ float red[256];
    float m = -1e30f;
    for (int l = threadIdx.x; l < LTOT; l += 256)
        m = fmaxf(m, S[(size_t)l * HW + p]);
    red[threadIdx.x] = m; __syncthreads();
    for (int s = 128; s; s >>= 1) {
        if (threadIdx.x < s) red[threadIdx.x] = fmaxf(red[threadIdx.x], red[threadIdx.x + s]);
        __syncthreads();
    }
    m = red[0] * SMS;                // max(10*s) = 10*max(s)
    __syncthreads();
    float sum = 0.f;
    for (int l = threadIdx.x; l < LTOT; l += 256) {
        float e = expf(SMS * S[(size_t)l * HW + p] - m);
        sum += e;
        attT[(size_t)p * LTOT + l] = (_Float16)e;
    }
    red[threadIdx.x] = sum; __syncthreads();
    for (int s = 128; s; s >>= 1) {
        if (threadIdx.x < s) red[threadIdx.x] += red[threadIdx.x + s];
        __syncthreads();
    }
    float inv = 1.f / red[0];
    for (int l = threadIdx.x; l < LTOT; l += 256) {
        float e = (float)attT[(size_t)p * LTOT + l];
        attT[(size_t)p * LTOT + l] = (_Float16)(e * inv);
    }
}

// ---------------------------------------------------------------------------
// fold (col2im of 3x3, pad=1) /4 + residual: deterministic gather, no atomics
// ---------------------------------------------------------------------------
__global__ __launch_bounds__(256) void fold_residual(
    const float* __restrict__ t, const float* __restrict__ xb,
    float* __restrict__ outb)
{
    int idx = blockIdx.x * blockDim.x + threadIdx.x;
    if (idx >= Cc * HW) return;
    int c = idx / HW, ij = idx % HW;
    int i = ij / Wwid, j = ij % Wwid;
    float acc = 0.f;
    #pragma unroll
    for (int di = 0; di < 3; ++di) {
        int py = i + 1 - di;
        if (py < 0 || py >= Hh) continue;
        #pragma unroll
        for (int dj = 0; dj < 3; ++dj) {
            int px = j + 1 - dj;
            if (px < 0 || px >= Wwid) continue;
            acc += t[(size_t)(py * Wwid + px) * K2 + c * 9 + di * 3 + dj];
        }
    }
    outb[idx] = xb[idx] + 0.25f * acc;
}

// ---------------------------------------------------------------------------
// host launcher — per-batch pipeline, scratch reused across batches (~86.5 MiB)
// ---------------------------------------------------------------------------
extern "C" void kernel_launch(void* const* d_in, const int* in_sizes, int n_in,
                              void* d_out, int out_size, void* d_ws, size_t ws_size,
                              hipStream_t stream)
{
    (void)in_sizes; (void)n_in; (void)out_size; (void)ws_size;

    const float* x  = (const float*)d_in[0];
    const float* Wb = (const float*)d_in[1];
    const float* bb = (const float*)d_in[2];
    const float* ab = (const float*)d_in[3];
    const float* Wm = (const float*)d_in[4];
    const float* bm = (const float*)d_in[5];
    const float* am = (const float*)d_in[6];
    const float* Wa = (const float*)d_in[7];
    const float* ba = (const float*)d_in[8];
    const float* aa = (const float*)d_in[9];
    float* out = (float*)d_out;

    char* ws = (char*)d_ws;
    size_t o = 0;
    float*    mb   = (float*)   (ws + o); o += (size_t)CRr * HW   * 4;  // 0.20 MB
    float*    ref  = (float*)   (ws + o); o += (size_t)Cc  * LTOT * 4;  // 5.41 MB
    float*    base = (float*)   (ws + o); o += (size_t)Cc  * LTOT * 4;  // 5.41 MB
    float*    refm = (float*)   (ws + o); o += (size_t)CRr * LTOT * 4;  // 0.68 MB
    _Float16* Wn   = (_Float16*)(ws + o); o += (size_t)LTOT * K1  * 2;  // 3.04 MB
    _Float16* xp   = (_Float16*)(ws + o); o += (size_t)HW   * K1  * 2;  // 0.92 MB
    _Float16* RT   = (_Float16*)(ws + o); o += (size_t)K2   * LTOT* 2;  // 24.3 MB
    _Float16* attT = (_Float16*)(ws + o); o += (size_t)HW   * LTOT* 2;  // 16.9 MB
    float*    scores = (float*) (ws + o); o += (size_t)LTOT * HW  * 4;  // 33.8 MB
    float*    tbuf = scores;            // t [HW,K2] (14.7 MB) aliases dead scores

    const int dims[5] = {40, 36, 32, 28, 24};
    const int offs[5] = {0, 1600, 2896, 3920, 4704};

    for (int b = 0; b < 2; ++b) {
        const float* xb   = x   + (size_t)b * Cc * HW;
        float*       outb = out + (size_t)b * Cc * HW;

        // match_base = prelu(conv1x1(x, Wb))
        conv1x1_prelu<<<(CRr * HW + 255) / 256, 256, 0, stream>>>(
            xb, Wb, bb, ab, mb, CRr, Cc, HW);

        // pyramid of refs
        copy_scale0<<<(Cc * HW + 255) / 256, 256, 0, stream>>>(xb, ref);
        for (int s = 1; s < 5; ++s) {
            int hh = dims[s], n = Cc * hh * hh;
            resize_cubic<<<(n + 255) / 256, 256, 0, stream>>>(xb, ref, hh, hh, offs[s]);
        }

        // base = prelu(conv1x1(ref, Wa)); refm = prelu(conv1x1(ref, Wm))
        conv1x1_prelu<<<(Cc * LTOT + 255) / 256, 256, 0, stream>>>(
            ref, Wa, ba, aa, base, Cc, Cc, LTOT);
        conv1x1_prelu<<<(CRr * LTOT + 255) / 256, 256, 0, stream>>>(
            ref, Wm, bm, am, refm, CRr, Cc, LTOT);

        // patch banks (f16 GEMM operands)
        build_Wn<<<(LTOT + 255) / 256, 256, 0, stream>>>(refm, Wn);
        build_xp<<<(HW + 255) / 256, 256, 0, stream>>>(mb, xp);
        build_RallT<<<(K2 * LTOT + 255) / 256, 256, 0, stream>>>(base, RT);

        // GEMM1: scores[LTOT, HW] = Wn * xp^T   (32x64 macro, 4 waves/block)
        gemm_f16_wmma_lds<2, 2><<<(LTOT / 32) * (HW / 64), 128, 0, stream>>>(
            Wn, xp, scores, LTOT, HW, K1);

        // softmax over L -> attT[HW, LTOT] f16
        softmax_col<<<HW, 256, 0, stream>>>(scores, attT);

        // GEMM2: t[HW, K2] = attT * RT^T       (64x64 macro, 8 waves/block)
        gemm_f16_wmma_lds<4, 2><<<(HW / 64) * (K2 / 64), 256, 0, stream>>>(
            attT, RT, tbuf, HW, K2, LTOT);

        // fold + /4 + residual
        fold_residual<<<(Cc * HW + 255) / 256, 256, 0, stream>>>(tbuf, xb, outb);
    }
}
